// EdgeConvBlock_74826920231269
// MI455X (gfx1250) — matrix operand order repaired and checked by reading, not compiled, not run on hardware
//
#include <hip/hip_runtime.h>
#include <hip/hip_bf16.h>

// ---------------------------------------------------------------------------
// EdgeConv block for MI455X (gfx1250, wave32, WMMA).
//   B=4, C=64, N=8192, k=20, c_out=64  (hardcoded per setup_inputs()).
// Phase 1: KNN via bf16 hi/lo-split WMMA Gram matrix (6x v_wmma_f32_16x16x32_bf16
//          per 16x16 distance tile), LDS insertion-sorted top-20.
// Phase 2: edge MLP via exact v_wmma_f32_16x16x4_f32; BN stats via LDS/global
//          float atomics; output max-over-k via integer-bit atomicMax (h>=0).
// ---------------------------------------------------------------------------

#define BB 4
#define CC 64
#define NN 8192
#define KK 20
#define COUT 64
#define ROWS (BB * NN * KK)      // 655360
#define NKP (NN * KK)            // rows per batch

typedef __attribute__((ext_vector_type(16))) __bf16 v16bf;
typedef __attribute__((ext_vector_type(8)))  __bf16 v8bf;
typedef __attribute__((ext_vector_type(8)))  float  v8f;
typedef __attribute__((ext_vector_type(2)))  float  v2f;

// ---- WMMA wrappers --------------------------------------------------------
__device__ __forceinline__ v8f wmma_bf16(v16bf a, v16bf b, v8f c) {
  return __builtin_amdgcn_wmma_f32_16x16x32_bf16(false, a, false, b,
                                                 (short)0, c, false, false);
}
__device__ __forceinline__ v8f wmma_f32(v2f a, v2f b, v8f c) {
  return __builtin_amdgcn_wmma_f32_16x16x4_f32(false, a, false, b,
                                               (short)0, c, false, false);
}

// A fragment (16x32 bf16, MxK): lane<16 -> koff 0, lane>=16 -> koff 16.
// halves 0..7  hold K = kc+koff+0..7 ; halves 8..15 hold K = kc+koff+8..15.
__device__ __forceinline__ v16bf ldA(const __bf16* row, int kc, int koff) {
  v8bf lo = *(const v8bf*)(row + kc + koff);
  v8bf hi = *(const v8bf*)(row + kc + koff + 8);
  v16bf r;
#pragma unroll
  for (int i = 0; i < 8; ++i) { r[i] = lo[i]; r[i + 8] = hi[i]; }
  return r;
}

// B fragment (32x16 bf16, KxN): lane<16 -> K = kc..kc+15, lane>=16 -> kc+16..kc+31,
// N = lane&15. 16 contiguous halves = one 32-byte load.
__device__ __forceinline__ v16bf ldB(const __bf16* row, int kc, int off16) {
  return *(const v16bf*)(row + kc + off16);
}

// ---- Kernel 1: transpose + bf16 hi/lo split + squared norms ---------------
// x: (B, C, N) f32 -> xtf: (B, N, C) f32, xhi/xlo: (B, N, C) bf16, sq: (B, N)
__global__ __launch_bounds__(256) void prep_kernel(const float* __restrict__ x,
                                                   float* __restrict__ xtf,
                                                   __bf16* __restrict__ xhi,
                                                   __bf16* __restrict__ xlo,
                                                   float* __restrict__ sq) {
  int t = blockIdx.x * blockDim.x + threadIdx.x;  // t in [0, B*N)
  if (t >= BB * NN) return;
  int b = t / NN, n = t % NN;
  const float* xb = x + (size_t)b * CC * NN;
  float* orow = xtf + (size_t)t * CC;
  __bf16* hrow = xhi + (size_t)t * CC;
  __bf16* lrow = xlo + (size_t)t * CC;
  float s = 0.f;
#pragma unroll
  for (int c = 0; c < CC; ++c) {
    float v = xb[c * NN + n];           // coalesced across lanes (consecutive n)
    orow[c] = v;
    __bf16 h = (__bf16)v;
    hrow[c] = h;
    lrow[c] = (__bf16)(v - (float)h);
    s += v * v;
  }
  sq[t] = s;
}

// ---- Kernel 2: KNN via WMMA + LDS top-20 ----------------------------------
// grid = B * (N/64) blocks, 128 threads (4 waves, 16 queries per wave).
__global__ __launch_bounds__(128) void knn_kernel(const __bf16* __restrict__ xhi,
                                                  const __bf16* __restrict__ xlo,
                                                  const float* __restrict__ sq,
                                                  int* __restrict__ idxout) {
  __shared__ float sdist[4][16][16];
  __shared__ float sbd[4][16][KK];
  __shared__ int   sbi[4][16][KK];

  const int tid = threadIdx.x;
  const int wave = tid >> 5;
  const int lane = tid & 31;
  const int ln = lane & 15;
  const bool hihalf = lane >= 16;

  const int b = blockIdx.x >> 7;                 // N/64 = 128 blocks per batch
  const int q0 = ((blockIdx.x & 127) << 6) + wave * 16;

  const __bf16* Hb = xhi + (size_t)b * NN * CC;
  const __bf16* Lb = xlo + (size_t)b * NN * CC;
  const float* sqb = sq + b * NN;

  if (!hihalf) {
#pragma unroll
    for (int j = 0; j < KK; ++j) { sbd[wave][ln][j] = __builtin_inff(); sbi[wave][ln][j] = 0; }
  }

  // Loop-invariant A fragments for this wave's 16 query rows (hi/lo, K-chunks 0 and 32).
  const int koff = hihalf ? 16 : 0;
  const __bf16* qH = Hb + (size_t)(q0 + ln) * CC;
  const __bf16* qL = Lb + (size_t)(q0 + ln) * CC;
  v16bf AH0 = ldA(qH, 0, koff),  AH1 = ldA(qH, 32, koff);
  v16bf AL0 = ldA(qL, 0, koff),  AL1 = ldA(qL, 32, koff);

  float sq_q[8];
#pragma unroll
  for (int r = 0; r < 8; ++r) sq_q[r] = sqb[q0 + r + (hihalf ? 8 : 0)];

  const int off16 = hihalf ? 16 : 0;

  for (int nc = 0; nc < NN; nc += 16) {
    const __bf16* cH = Hb + (size_t)(nc + ln) * CC;
    const __bf16* cL = Lb + (size_t)(nc + ln) * CC;
    v16bf BH0 = ldB(cH, 0, off16), BH1 = ldB(cH, 32, off16);
    v16bf BL0 = ldB(cL, 0, off16), BL1 = ldB(cL, 32, off16);

    v8f acc = {};
    acc = wmma_bf16(AH0, BH0, acc);   // hi*hi
    acc = wmma_bf16(AH0, BL0, acc);   // hi*lo
    acc = wmma_bf16(AL0, BH0, acc);   // lo*hi
    acc = wmma_bf16(AH1, BH1, acc);
    acc = wmma_bf16(AH1, BL1, acc);
    acc = wmma_bf16(AL1, BH1, acc);

    const int cand = nc + ln;
    const float sqc = sqb[cand];
#pragma unroll
    for (int r = 0; r < 8; ++r) {
      int qm = r + (hihalf ? 8 : 0);
      float d = sq_q[r] + sqc - 2.f * acc[r];
      if (q0 + qm == cand) d = __builtin_inff();   // self mask
      sdist[wave][qm][ln] = d;
    }
    __builtin_amdgcn_wave_barrier();

    if (!hihalf) {
      float dmax = sbd[wave][ln][KK - 1];
#pragma unroll 4
      for (int j = 0; j < 16; ++j) {
        float d = sdist[wave][ln][j];
        if (d < dmax) {
          int p = KK - 1;
          while (p > 0 && sbd[wave][ln][p - 1] > d) {
            sbd[wave][ln][p] = sbd[wave][ln][p - 1];
            sbi[wave][ln][p] = sbi[wave][ln][p - 1];
            --p;
          }
          sbd[wave][ln][p] = d;
          sbi[wave][ln][p] = nc + j;
          dmax = sbd[wave][ln][KK - 1];
        }
      }
    }
    __builtin_amdgcn_wave_barrier();
  }

  if (!hihalf) {
    int* dst = idxout + ((size_t)(b * NN) + (q0 + ln)) * KK;
#pragma unroll
    for (int j = 0; j < KK; ++j) dst[j] = sbi[wave][ln][j];
  }
}

// ---- Kernel 3: zero output + stat accumulators ----------------------------
__global__ __launch_bounds__(256) void init_kernel(float* __restrict__ out, int n,
                                                   float* __restrict__ gsum,
                                                   float* __restrict__ gsumsq) {
  int t = blockIdx.x * blockDim.x + threadIdx.x;
  if (t < n) out[t] = 0.f;
  if (t < COUT) { gsum[t] = 0.f; gsumsq[t] = 0.f; }
}

// ---- Kernels 4/6: edge MLP via f32 WMMA; MODE 0 = stats, MODE 1 = output --
// grid = ROWS/64 blocks, 128 threads (4 waves, 16 rows each).
template <int MODE>
__global__ __launch_bounds__(128) void edge_kernel(const float* __restrict__ xtf,
                                                   const int* __restrict__ idx,
                                                   const float* __restrict__ W,
                                                   float* __restrict__ gsum,
                                                   float* __restrict__ gsumsq,
                                                   const float* __restrict__ scale,
                                                   const float* __restrict__ shift,
                                                   float* __restrict__ out) {
  __shared__ float sW[COUT * 2 * CC];   // 32 KB: W staged once per WG
  __shared__ float sE[4][16][2 * CC];   // 32 KB: per-wave edge tile
  __shared__ float wsum[COUT], wsum2[COUT];

  const int tid = threadIdx.x;
  const int wave = tid >> 5;
  const int lane = tid & 31;
  const int ln = lane & 15;
  const bool hihalf = lane >= 16;

  for (int i = tid; i < COUT * 2 * CC; i += 128) sW[i] = W[i];
  if (MODE == 0 && tid < COUT) { wsum[tid] = 0.f; wsum2[tid] = 0.f; }
  __syncthreads();

  const int r0 = (blockIdx.x * 4 + wave) * 16;

  // Build 16 edge rows [nbr-center | center] in LDS (cooperative per wave).
  for (int rr = 0; rr < 16; ++rr) {
    int r = r0 + rr;
    int b = r / NKP;
    int rem = r % NKP;
    int n = rem / KK, j = rem % KK;
    int nbr = idx[(size_t)(b * NN + n) * KK + j];
    const float* cen = xtf + (size_t)(b * NN + n) * CC;
    const float* ngb = xtf + (size_t)(b * NN + nbr) * CC;
    for (int c = lane; c < CC; c += 32) {
      float cv = cen[c];
      sE[wave][rr][c] = ngb[c] - cv;
      sE[wave][rr][CC + c] = cv;
    }
  }
  __syncthreads();

  const int off2 = hihalf ? 2 : 0;
#pragma unroll
  for (int nt = 0; nt < 4; ++nt) {
    v8f acc = {};
#pragma unroll
    for (int ks = 0; ks < 32; ++ks) {          // K = 128 in steps of 4
      int kb = ks * 4 + off2;
      v2f a, bb;
      a.x = sE[wave][ln][kb];
      a.y = sE[wave][ln][kb + 1];
      bb.x = sW[(nt * 16 + ln) * 2 * CC + kb];
      bb.y = sW[(nt * 16 + ln) * 2 * CC + kb + 1];
      acc = wmma_f32(a, bb, acc);
    }
    const int o = nt * 16 + ln;
    if (MODE == 0) {
      float s = 0.f, s2 = 0.f;
#pragma unroll
      for (int r8 = 0; r8 < 8; ++r8) { s += acc[r8]; s2 += acc[r8] * acc[r8]; }
      atomicAdd(&wsum[o], s);                  // ds_add_f32
      atomicAdd(&wsum2[o], s2);
    } else {
      const float sc = scale[o], sh = shift[o];
#pragma unroll
      for (int r8 = 0; r8 < 8; ++r8) {
        float h = acc[r8] * sc + sh;
        h = h > 0.f ? h : 0.f;                 // ReLU -> h >= 0
        int r = r0 + r8 + (hihalf ? 8 : 0);
        int b = r / NKP;
        int n = (r % NKP) / KK;
        // max over k via integer-bit max (monotonic for nonnegative floats)
        atomicMax((int*)&out[((size_t)b * COUT + o) * NN + n], __float_as_int(h));
      }
    }
  }

  if (MODE == 0) {
    __syncthreads();
    if (tid < COUT) {
      atomicAdd(&gsum[tid], wsum[tid]);
      atomicAdd(&gsumsq[tid], wsum2[tid]);
    }
  }
}

// ---- Kernel 5: BN finalize: fold mean/var/gamma/beta into scale/shift -----
__global__ __launch_bounds__(64) void bn_finalize_kernel(const float* __restrict__ gsum,
                                                         const float* __restrict__ gsumsq,
                                                         const float* __restrict__ gamma,
                                                         const float* __restrict__ beta,
                                                         float* __restrict__ scale,
                                                         float* __restrict__ shift) {
  int o = threadIdx.x;
  if (o >= COUT) return;
  const float cnt = (float)ROWS;
  float mean = gsum[o] / cnt;
  float var = gsumsq[o] / cnt - mean * mean;
  float inv = rsqrtf(var + 1e-5f);
  float sc = gamma[o] * inv;
  scale[o] = sc;
  shift[o] = beta[o] - mean * sc;
}

// ---------------------------------------------------------------------------
extern "C" void kernel_launch(void* const* d_in, const int* in_sizes, int n_in,
                              void* d_out, int out_size, void* d_ws, size_t ws_size,
                              hipStream_t stream) {
  const float* x     = (const float*)d_in[0];   // (B, C, N)
  const float* W     = (const float*)d_in[1];   // (64, 128)
  const float* gamma = (const float*)d_in[2];
  const float* beta  = (const float*)d_in[3];
  float* out = (float*)d_out;                   // (B, 64, N)

  char* ws = (char*)d_ws;
  size_t off = 0;
  auto alloc = [&](size_t bytes) -> void* {
    void* p = ws + off;
    off = (off + bytes + 255) & ~(size_t)255;
    return p;
  };
  float*  xtf    = (float*)alloc((size_t)BB * NN * CC * sizeof(float));
  __bf16* xhi    = (__bf16*)alloc((size_t)BB * NN * CC * sizeof(__bf16));
  __bf16* xlo    = (__bf16*)alloc((size_t)BB * NN * CC * sizeof(__bf16));
  float*  sq     = (float*)alloc((size_t)BB * NN * sizeof(float));
  int*    idx    = (int*)alloc((size_t)BB * NN * KK * sizeof(int));
  float*  gsum   = (float*)alloc(COUT * sizeof(float));
  float*  gsumsq = (float*)alloc(COUT * sizeof(float));
  float*  scale  = (float*)alloc(COUT * sizeof(float));
  float*  shift  = (float*)alloc(COUT * sizeof(float));

  // 1) transpose + bf16 hi/lo split + norms
  prep_kernel<<<(BB * NN + 255) / 256, 256, 0, stream>>>(x, xtf, xhi, xlo, sq);

  // 2) KNN top-20
  knn_kernel<<<BB * (NN / 64), 128, 0, stream>>>(xhi, xlo, sq, idx);

  // 3) zero output + accumulators
  init_kernel<<<(out_size + 255) / 256, 256, 0, stream>>>(out, out_size, gsum, gsumsq);

  // 4) BN statistics pass
  edge_kernel<0><<<ROWS / 64, 128, 0, stream>>>(xtf, idx, W, gsum, gsumsq,
                                                scale, shift, out);

  // 5) fold BN params
  bn_finalize_kernel<<<1, 64, 0, stream>>>(gsum, gsumsq, gamma, beta, scale, shift);

  // 6) output pass: recompute h, BN + ReLU + max over k
  edge_kernel<1><<<ROWS / 64, 128, 0, stream>>>(xtf, idx, W, gsum, gsumsq,
                                                scale, shift, out);
}
